// GATJKClassifier_5772436046309
// MI455X (gfx1250) — matrix operand order repaired and checked by reading, not compiled, not run on hardware
//
#include <hip/hip_runtime.h>
#include <hip/hip_bf16.h>

// ---------------------------------------------------------------------------
// GATv2 + JumpingKnowledge classifier for MI455X (gfx1250, wave32, WMMA).
// Dense GEMMs: operands pre-packed to bf16 (weights transposed so fragments
// are contiguous), inner loop is pure b128 loads + v_wmma_f32_16x16x32_bf16.
// Edge softmax / scatter phases use L2-resident float atomics.
// ---------------------------------------------------------------------------

typedef __attribute__((ext_vector_type(16))) __bf16          v16bf;
typedef __attribute__((ext_vector_type(16))) unsigned short  v16us;
typedef __attribute__((ext_vector_type(8)))  unsigned short  v8us;
typedef __attribute__((ext_vector_type(8)))  float           v8f;

#define HCc 64          // hidden channels per head
#define WAVE 32

__device__ __forceinline__ unsigned short f2bf(float f) {
  unsigned u = __builtin_bit_cast(unsigned, f);
  return (unsigned short)((u + 0x7FFFu + ((u >> 16) & 1u)) >> 16);   // RNE
}

// monotonic order-preserving float -> unsigned key (for atomicMax)
__device__ __forceinline__ unsigned fkey(float f) {
  unsigned u = __builtin_bit_cast(unsigned, f);
  return u ^ ((u >> 31) ? 0xFFFFFFFFu : 0x80000000u);
}
__device__ __forceinline__ float funkey(unsigned k) {
  k ^= (k >> 31) ? 0x80000000u : 0xFFFFFFFFu;
  return __builtin_bit_cast(float, k);
}

// ---- operand packing -------------------------------------------------------
// row-major f32 -> row-major bf16
__global__ void pack_bf16(const float* __restrict__ in, unsigned short* __restrict__ out,
                          size_t n) {
  size_t i = (size_t)blockIdx.x * blockDim.x + threadIdx.x;
  if (i < n) out[i] = f2bf(in[i]);
}
// W[K,Nc] f32 -> Wt[Nc,K] bf16 (coalesced writes, strided L2-hot reads)
__global__ void pack_wt(const float* __restrict__ W, unsigned short* __restrict__ out,
                        int K, int Nc) {
  int idx = blockIdx.x * blockDim.x + threadIdx.x;
  if (idx >= K * Nc) return;
  int n = idx / K, k = idx - n * K;
  out[idx] = f2bf(W[(size_t)k * Nc + n]);
}

// ---------------------------------------------------------------------------
// C[M,Nc] = Abf[M,K] @ WbfT[Nc,K]^T + bias.  bf16 in, f32 WMMA accumulate.
// One wave -> 16x64 output tile (4 accumulators). M%16==0, K%32==0, Nc%64==0.
// ---------------------------------------------------------------------------
__global__ void gemm_bf16_wmma(const unsigned short* __restrict__ Abf,
                               const unsigned short* __restrict__ WbfT,
                               const float* __restrict__ bias,
                               float* __restrict__ C,
                               int M, int K, int Nc) {
  const int lane   = threadIdx.x & (WAVE - 1);
  const int wave   = blockIdx.x * (blockDim.x >> 5) + (threadIdx.x >> 5);
  const int tilesN = Nc >> 6;
  const int tilesM = M >> 4;
  if (wave >= tilesM * tilesN) return;
  const int m0 = (wave / tilesN) << 4;
  const int n0 = (wave % tilesN) << 6;

  v8f acc[4] = {v8f{}, v8f{}, v8f{}, v8f{}};

  const int half = (lane < 16) ? 0 : 1;
  const int l15  = lane & 15;
  const int akb  = half ? 16 : 0;      // A frag K base within 32-step
  const int bkb  = half ? 16 : 0;      // B frag K base within 32-step
  const unsigned short* arow  = Abf + (size_t)(m0 + l15) * K;
  const unsigned short* wcol0 = WbfT + (size_t)(n0 + l15) * K + bkb;

  for (int k0 = 0; k0 < K; k0 += 32) {
    __builtin_prefetch(arow + k0 + 128, 0, 0);        // global_prefetch_b8
    // A fragment: lanes0-15 hold K{0-15}, lanes16-31 K{16-31}
    v8us a0 = *(const v8us*)(arow + k0 + akb);
    v8us a1 = *(const v8us*)(arow + k0 + akb + 8);
    v16bf a = __builtin_bit_cast(v16bf,
        __builtin_shufflevector(a0, a1, 0,1,2,3,4,5,6,7,8,9,10,11,12,13,14,15));
#pragma unroll
    for (int nt = 0; nt < 4; ++nt) {
      // B fragment: column n0+nt*16+l15, 16 consecutive K -> 32 contiguous B
      v16us bu = *(const v16us*)(wcol0 + (size_t)nt * 16 * K + k0);
      v16bf b  = __builtin_bit_cast(v16bf, bu);
      acc[nt] = __builtin_amdgcn_wmma_f32_16x16x32_bf16(
          false, a, false, b, (short)0, acc[nt], false, false);
    }
  }

  // store: C/D layout (lane<16: rows m0+r, lane>=16: rows m0+8+r)
  const int rbase = half ? 8 : 0;
#pragma unroll
  for (int nt = 0; nt < 4; ++nt) {
    const int col = n0 + nt * 16 + l15;
    const float bv = bias ? bias[col] : 0.f;
#pragma unroll
    for (int r = 0; r < 8; ++r)
      C[(size_t)(m0 + rbase + r) * Nc + col] = acc[nt][r] + bv;
  }
}

// ---------------------------------------------------------------------------
__global__ void zero_buf(float* __restrict__ p, size_t n) {
  size_t i = (size_t)blockIdx.x * blockDim.x + threadIdx.x;
  if (i < n) p[i] = 0.f;
}

__device__ __forceinline__ void edge_sd(const int* ei, int E, int e, int& s, int& d) {
  if (e < E) { s = ei[e]; d = ei[E + e]; } else { s = d = e - E; }
}

// logit[e,h] = att_h . leaky_relu(xl[src,h,:] + xr[dst,h,:]); atomicMax key
__global__ void edge_logits(const float* __restrict__ xl, const float* __restrict__ xr,
                            const float* __restrict__ att, const int* __restrict__ ei,
                            int E, int ET, int heads,
                            float* __restrict__ logit, unsigned* __restrict__ mkey) {
  int idx = blockIdx.x * blockDim.x + threadIdx.x;
  if (idx >= ET * heads) return;
  int e = idx / heads, h = idx - e * heads;
  int s, d; edge_sd(ei, E, e, s, d);
  const int D = heads * HCc;
  const float4* pl = (const float4*)(xl + (size_t)s * D + h * HCc);
  const float4* pr = (const float4*)(xr + (size_t)d * D + h * HCc);
  const float4* pa = (const float4*)(att + h * HCc);
  float acc = 0.f;
#pragma unroll
  for (int c = 0; c < HCc / 4; ++c) {
    float4 a = pl[c], b = pr[c], w = pa[c];
    float v0 = a.x + b.x, v1 = a.y + b.y, v2 = a.z + b.z, v3 = a.w + b.w;
    v0 = v0 > 0.f ? v0 : 0.2f * v0;  v1 = v1 > 0.f ? v1 : 0.2f * v1;
    v2 = v2 > 0.f ? v2 : 0.2f * v2;  v3 = v3 > 0.f ? v3 : 0.2f * v3;
    acc += w.x * v0 + w.y * v1 + w.z * v2 + w.w * v3;
  }
  logit[idx] = acc;
  atomicMax(mkey + (size_t)d * heads + h, fkey(acc));
}

// p = exp(logit - max); atomicAdd segment sums (p overwrites logit in place)
__global__ void edge_p(float* __restrict__ logit, const unsigned* __restrict__ mkey,
                       const int* __restrict__ ei, int E, int ET, int heads,
                       float* __restrict__ ssum) {
  int idx = blockIdx.x * blockDim.x + threadIdx.x;
  if (idx >= ET * heads) return;
  int e = idx / heads, h = idx - e * heads;
  int s, d; edge_sd(ei, E, e, s, d); (void)s;
  float m = funkey(mkey[(size_t)d * heads + h]);
  float p = __expf(logit[idx] - m);
  logit[idx] = p;
  atomicAdd(ssum + (size_t)d * heads + h, p);
}

// out[dst,h,:] += (p / ssum[dst,h]) * xl[src,h,:]
__global__ void edge_accum(const float* __restrict__ p, const float* __restrict__ ssum,
                           const float* __restrict__ xl, const int* __restrict__ ei,
                           int E, int ET, int heads, float* __restrict__ out) {
  int idx = blockIdx.x * blockDim.x + threadIdx.x;
  if (idx >= ET * heads) return;
  int e = idx / heads, h = idx - e * heads;
  int s, d; edge_sd(ei, E, e, s, d);
  const int D = heads * HCc;
  float alpha = p[idx] / ssum[(size_t)d * heads + h];
  const float4* pl = (const float4*)(xl + (size_t)s * D + h * HCc);
  float* po = out + (size_t)d * D + h * HCc;
#pragma unroll 4
  for (int c = 0; c < HCc / 4; ++c) {
    float4 v = pl[c];
    atomicAdd(po + 4 * c + 0, alpha * v.x);
    atomicAdd(po + 4 * c + 1, alpha * v.y);
    atomicAdd(po + 4 * c + 2, alpha * v.z);
    atomicAdd(po + 4 * c + 3, alpha * v.w);
  }
}

// out += bias (+ optional residual)
__global__ void bias_res(float* __restrict__ out, const float* __restrict__ bias,
                         const float* __restrict__ res, size_t n, int D) {
  size_t idx = (size_t)blockIdx.x * blockDim.x + threadIdx.x;
  if (idx >= n) return;
  int d = (int)(idx % D);
  float v = out[idx] + bias[d];
  if (res) v += res[idx];
  out[idx] = v;
}

// per-(graph, channel) GraphNorm stats; block = (1 graph, D threads)
__global__ void gnorm_stats(const float* __restrict__ x, const float* __restrict__ ms,
                            float* __restrict__ mean, float* __restrict__ istd,
                            int D, int npg) {
  int g = blockIdx.x, d = threadIdx.x;
  const float* base = x + (size_t)g * npg * D + d;
  float s = 0.f;
  for (int n = 0; n < npg; ++n) s += base[(size_t)n * D];
  float mu = s / (float)npg;
  float msd = ms[d];
  float q = 0.f;
  for (int n = 0; n < npg; ++n) { float c = base[(size_t)n * D] - msd * mu; q += c * c; }
  mean[(size_t)g * D + d] = mu;
  istd[(size_t)g * D + d] = rsqrtf(q / (float)npg + 1e-5f);
}

__global__ void gnorm_apply(const float* __restrict__ x, const float* __restrict__ mean,
                            const float* __restrict__ istd, const float* __restrict__ ms,
                            const float* __restrict__ w, const float* __restrict__ b,
                            float* __restrict__ out, size_t n, int D, int npg, int doElu) {
  size_t idx = (size_t)blockIdx.x * blockDim.x + threadIdx.x;
  if (idx >= n) return;
  int d = (int)(idx % D);
  int g = (int)((idx / D) / npg);
  float c = x[idx] - ms[d] * mean[(size_t)g * D + d];
  float y = w[d] * c * istd[(size_t)g * D + d] + b[d];
  if (doElu) y = y > 0.f ? y : __expf(y) - 1.f;
  out[idx] = y;
}

// copy JK summary rows (node g*npg) into [G, FD] at column offset `off`
__global__ void copy_summary(const float* __restrict__ h, int D, int off,
                             float* __restrict__ sumb, int npg, int FD, int G) {
  int idx = blockIdx.x * blockDim.x + threadIdx.x;
  if (idx >= G * D) return;
  int g = idx / D, d = idx - g * D;
  sumb[(size_t)g * FD + off + d] = h[(size_t)g * npg * D + d];
}

__global__ void final_lin(const float* __restrict__ sumb, const float* __restrict__ lw,
                          const float* __restrict__ lb, float* __restrict__ out,
                          int G, int FD) {
  int g = blockIdx.x * blockDim.x + threadIdx.x;
  if (g >= G) return;
  float acc = lb[0];
  for (int f = 0; f < FD; ++f) acc += sumb[(size_t)g * FD + f] * lw[f];
  out[g] = acc;
}

// ---------------------------------------------------------------------------
extern "C" void kernel_launch(void* const* d_in, const int* in_sizes, int n_in,
                              void* d_out, int out_size, void* d_ws, size_t ws_size,
                              hipStream_t stream) {
  (void)n_in; (void)out_size; (void)ws_size;
  const float* x     = (const float*)d_in[0];
  const int*   ei    = (const int*)d_in[1];
  const float* tp_w  = (const float*)d_in[4];
  const float* tp_b  = (const float*)d_in[5];
  const float* lin_w = (const float*)d_in[6];
  const float* lin_b = (const float*)d_in[7];
  const float* cw[4][6];   // wl, bl, wr, br, att, bias
  for (int l = 0; l < 4; ++l)
    for (int j = 0; j < 6; ++j) cw[l][j] = (const float*)d_in[8 + l * 6 + j];
  const float* nw[4][3];   // w, b, ms
  for (int l = 0; l < 4; ++l)
    for (int j = 0; j < 3; ++j) nw[l][j] = (const float*)d_in[32 + l * 3 + j];

  const int N   = in_sizes[2];          // 20000
  const int TD  = in_sizes[0] / N;      // 768
  const int E   = in_sizes[1] / 2;      // 240000
  const int G   = in_sizes[3] - 1;      // 50
  const int NPG = N / G;                // 400
  const int ET  = E + N;                // + self loops
  const int H   = 8;
  const int FD  = 3 * H * HCc + HCc;    // 1600

  // ---- workspace layout (256B aligned) ----
  char* base = (char*)d_ws;
  size_t off = 0;
  auto takeb = [&](size_t bytes) -> void* {
    off = (off + 255) & ~(size_t)255;
    void* p = (void*)(base + off);
    off += bytes;
    return p;
  };
  auto take = [&](size_t floats) -> float* { return (float*)takeb(floats * 4); };
  float*    XL   = take((size_t)N * 512);
  float*    XR   = take((size_t)N * 512);
  float*    Hb   = take((size_t)N * 512);   // current activation h
  float*    Cb   = take((size_t)N * 512);   // conv output / pre-norm
  float*    LOG  = take((size_t)ET * H);    // logits -> p
  unsigned* MKEY = (unsigned*)take((size_t)N * H);
  float*    SSUM = take((size_t)N * H);
  float*    MEAN = take((size_t)G * 512);
  float*    ISTD = take((size_t)G * 512);
  float*    SUMB = take((size_t)G * FD);
  unsigned short* Abf = (unsigned short*)takeb((size_t)N * TD * 2);   // packed A
  unsigned short* Wbf = (unsigned short*)takeb((size_t)512 * 768 * 2); // packed W^T

  auto zero = [&](float* p, size_t n) {
    zero_buf<<<(int)((n + 1023) / 1024), 1024, 0, stream>>>(p, n);
  };
  auto packA = [&](const float* A, size_t n) {
    pack_bf16<<<(int)((n + 1023) / 1024), 1024, 0, stream>>>(A, Abf, n);
  };
  auto gemm = [&](const float* W, const float* bias, float* C, int M, int K, int Nc) {
    pack_wt<<<(K * Nc + 255) / 256, 256, 0, stream>>>(W, Wbf, K, Nc);
    int waves = (M / 16) * (Nc / 64);
    gemm_bf16_wmma<<<(waves + 7) / 8, 256, 0, stream>>>(Abf, Wbf, bias, C, M, K, Nc);
  };

  // ---- text projection: Hb = x @ tp_w + tp_b  -> [N, 64] ----
  packA(x, (size_t)N * TD);
  gemm(tp_w, tp_b, Hb, N, TD, HCc);

  const int sumoff[4] = {0, 512, 1024, 1536};
  for (int l = 0; l < 4; ++l) {
    const int Din   = (l == 0) ? HCc : H * HCc;
    const int heads = (l == 3) ? 1 : H;
    const int Dout  = heads * HCc;
    const bool residual = (l == 1 || l == 2);
    const bool elu      = (l < 3);

    // xl = h @ wl + bl ; xr = h @ wr + br  (A packed once, reused)
    packA(Hb, (size_t)N * Din);
    gemm(cw[l][0], cw[l][1], XL, N, Din, Dout);
    gemm(cw[l][2], cw[l][3], XR, N, Din, Dout);

    zero((float*)MKEY, (size_t)N * heads);
    zero(SSUM,         (size_t)N * heads);
    zero(Cb,           (size_t)N * Dout);

    int eth = ET * heads;
    int ebl = (eth + 255) / 256;
    edge_logits<<<ebl, 256, 0, stream>>>(XL, XR, cw[l][4], ei, E, ET, heads, LOG, MKEY);
    edge_p     <<<ebl, 256, 0, stream>>>(LOG, MKEY, ei, E, ET, heads, SSUM);
    edge_accum <<<ebl, 256, 0, stream>>>(LOG, SSUM, XL, ei, E, ET, heads, Cb);

    size_t nd = (size_t)N * Dout;
    bias_res<<<(int)((nd + 255) / 256), 256, 0, stream>>>(
        Cb, cw[l][5], residual ? Hb : nullptr, nd, Dout);

    gnorm_stats<<<G, Dout, 0, stream>>>(Cb, nw[l][2], MEAN, ISTD, Dout, NPG);
    float* dest = (l < 3) ? Hb : Cb;
    gnorm_apply<<<(int)((nd + 255) / 256), 256, 0, stream>>>(
        Cb, MEAN, ISTD, nw[l][2], nw[l][0], nw[l][1], dest, nd, Dout, NPG, elu ? 1 : 0);

    copy_summary<<<(G * Dout + 255) / 256, 256, 0, stream>>>(
        dest, Dout, sumoff[l], SUMB, NPG, FD, G);
  }

  final_lin<<<1, 64, 0, stream>>>(SUMB, lin_w, lin_b, (float*)d_out, G, FD);
}
